// Grapher_73349451481284
// MI455X (gfx1250) — matrix operand order repaired and checked by reading, not compiled, not run on hardware
//
#include <hip/hip_runtime.h>
#include <hip/hip_bf16.h>

typedef __bf16 bf16_t;
typedef __attribute__((ext_vector_type(16))) __bf16 v16bf;
typedef __attribute__((ext_vector_type(8)))  float  v8f;

struct frag_bytes { uint4 lo; uint4 hi; };   // 32 bytes == v16bf

// ---------- helpers ----------
static __device__ __forceinline__ bf16_t f2bf(float f) {
  unsigned u = __builtin_bit_cast(unsigned, f);
  unsigned r = (u + 0x7FFFu + ((u >> 16) & 1u)) >> 16;   // RNE
  return __builtin_bit_cast(bf16_t, (unsigned short)r);
}
static __device__ __forceinline__ float lrelu(float x) { return x > 0.f ? x : 0.2f * x; }

// 2x b128 loads -> one 16-bf16 WMMA fragment (both pointers must be 16B aligned)
static __device__ __forceinline__ v16bf load_frag(const bf16_t* p0, const bf16_t* p1) {
  frag_bytes fb;
  fb.lo = *(const uint4*)p0;
  fb.hi = *(const uint4*)p1;
  return __builtin_bit_cast(v16bf, fb);
}

// float atomic max via int/uint trick (works with -inf init)
static __device__ __forceinline__ void atomicMaxF(float* a, float v) {
  if (v >= 0.f) atomicMax((int*)a, __builtin_bit_cast(int, v));
  else          atomicMin((unsigned*)a, __builtin_bit_cast(unsigned, v));
}

static __device__ __forceinline__ void edge_sd(const int* __restrict__ ei, int E, int e,
                                               int& s, int& d) {
  if (e < E) { s = ei[e]; d = ei[E + e]; }
  else       { s = e - E; d = e - E; }            // self loops appended
}

// ---------- transpose+convert: src[K x C] f32 -> dst[C x K] bf16 (coalesced both ways) ----------
__global__ void k_transpose_bf16(const float* __restrict__ src, bf16_t* __restrict__ dst,
                                 int K, int C) {
  __shared__ bf16_t tile[32][33];
  const int c0 = blockIdx.x * 32;
  const int k0 = blockIdx.y * 32;
  const int tx = threadIdx.x & 31;
  const int ty = threadIdx.x >> 5;               // 0..7
  #pragma unroll
  for (int i = ty; i < 32; i += 8) {
    int k = k0 + i, c = c0 + tx;
    tile[i][tx] = (k < K && c < C) ? f2bf(src[(long)k * C + c]) : f2bf(0.f);
  }
  __syncthreads();
  #pragma unroll
  for (int i = ty; i < 32; i += 8) {
    int c = c0 + i, k = k0 + tx;
    if (c < C && k < K) dst[(long)c * K + k] = tile[tx][i];
  }
}

// ---------- misc small kernels ----------
__global__ void k_embed(const int* __restrict__ nodes, const float* __restrict__ emb,
                        bf16_t* __restrict__ xbf, int N) {
  long idx = (long)blockIdx.x * blockDim.x + threadIdx.x;
  if (idx >= (long)N * 128) return;
  int i = (int)(idx >> 7), h = (int)(idx & 127);
  xbf[idx] = f2bf(emb[(long)nodes[i] * 128 + h]);
}

__global__ void k_init_layer(float* __restrict__ m, float* __restrict__ dn,
                             float* __restrict__ acc, int N) {
  long idx = (long)blockIdx.x * blockDim.x + threadIdx.x;
  if (idx < (long)N * 128) acc[idx] = 0.f;
  if (idx < N) { m[idx] = -__builtin_huge_valf(); dn[idx] = 0.f; }
}

// ---------- WMMA GEMM: C[M x ncols] = A[M x 128](bf16,row-major) * Bt[ncols x 128](bf16,K-contig) ----------
// one wave per 16x16 tile; 8 waves/block -> 16x128 C panel; K = 128 = 4 x wmma_16x16x32_bf16
// Requirements: M % 16 == 0, ncols % 128 == 0 (all call sites satisfy this).
template <bool BIAS, bool OUTF, bool OUTBF>
__global__ void k_gemm_bf16(const bf16_t* __restrict__ A, const bf16_t* __restrict__ Bt,
                            const float* __restrict__ bias,
                            float* __restrict__ outF, bf16_t* __restrict__ outBF,
                            int M, int ncols) {
  const int lane = threadIdx.x & 31;
  const int wave = threadIdx.x >> 5;
  const int row0 = blockIdx.x * 16;
  const int col0 = blockIdx.y * 128 + wave * 16;
  const int half = lane >> 4;
  const int l15  = lane & 15;

  v8f c = {0.f, 0.f, 0.f, 0.f, 0.f, 0.f, 0.f, 0.f};

  const bf16_t* Ar = A  + (long)(row0 + l15) * 128;  // 256B-aligned row
  const bf16_t* Br = Bt + (long)(col0 + l15) * 128;  // 256B-aligned row

  #pragma unroll
  for (int kb = 0; kb < 128; kb += 32) {
    // A fragment: VGPR0-3 = K[8h .. 8h+7], VGPR4-7 = K[16+8h .. 16+8h+7]  (two 16B segments)
    v16bf a = load_frag(Ar + kb + 8 * half, Ar + kb + 16 + 8 * half);
    // B fragment: VGPR0-3 = K[8h .. 8h+7], VGPR4-7 = K[16+8h .. 16+8h+7]  (two 16B segments)
    v16bf b = load_frag(Br + kb + 8 * half, Br + kb + 16 + 8 * half);
    c = __builtin_amdgcn_wmma_f32_16x16x32_bf16(false, a, false, b, (short)0, c, false, false);
  }

  float bv = 0.f;
  if (BIAS) bv = bias[col0 + l15];
  #pragma unroll
  for (int r = 0; r < 8; r++) {
    const int row = row0 + r + 8 * half;           // C layout: VGPR r -> M=r / M=8+r
    const long idx = (long)row * ncols + col0 + l15;
    const float v = c[r] + bv;
    if (OUTF)  outF[idx]  = v;
    if (OUTBF) outBF[idx] = f2bf(v);
  }
}

// ---------- per-node attention scores: e_src = xw . a_src, e_dst = xw . a_dst ----------
__global__ void k_escore(const float* __restrict__ xw, const float* __restrict__ as,
                         const float* __restrict__ ad, float* __restrict__ es,
                         float* __restrict__ ed, int N) {
  int node = blockIdx.x * (blockDim.x >> 5) + (threadIdx.x >> 5);
  if (node >= N) return;
  int lane = threadIdx.x & 31;
  const float* row = xw + (long)node * 128;
  float s = 0.f, d = 0.f;
  #pragma unroll
  for (int h = lane; h < 128; h += 32) {
    float v = row[h];
    s += v * as[h];
    d += v * ad[h];
  }
  #pragma unroll
  for (int off = 16; off > 0; off >>= 1) {
    s += __shfl_xor(s, off);
    d += __shfl_xor(d, off);
  }
  if (lane == 0) { es[node] = s; ed[node] = d; }
}

// ---------- segment softmax over incoming edges: 3 streaming passes ----------
__global__ void k_edge_max(const int* __restrict__ ei, int E, int EP,
                           const float* __restrict__ es, const float* __restrict__ ed,
                           float* __restrict__ m) {
  int e = blockIdx.x * blockDim.x + threadIdx.x;
  if (e >= EP) return;
  int s, d; edge_sd(ei, E, e, s, d);
  atomicMaxF(m + d, lrelu(es[s] + ed[d]));
}

__global__ void k_edge_sum(const int* __restrict__ ei, int E, int EP,
                           const float* __restrict__ es, const float* __restrict__ ed,
                           const float* __restrict__ m, float* __restrict__ dn) {
  int e = blockIdx.x * blockDim.x + threadIdx.x;
  if (e >= EP) return;
  int s, d; edge_sd(ei, E, e, s, d);
  atomicAdd(dn + d, __expf(lrelu(es[s] + ed[d]) - m[d]));
}

__global__ void k_edge_scatter(const int* __restrict__ ei, int E, int EP,
                               const float* __restrict__ es, const float* __restrict__ ed,
                               const float* __restrict__ m, const float* __restrict__ dn,
                               const float* __restrict__ xw, float* __restrict__ acc) {
  int e = blockIdx.x * (blockDim.x >> 5) + (threadIdx.x >> 5);
  if (e >= EP) return;
  int lane = threadIdx.x & 31;
  int s, d; edge_sd(ei, E, e, s, d);
  float w = __expf(lrelu(es[s] + ed[d]) - m[d]) / dn[d];
  const float* xr = xw + (long)s * 128;
  float* ar = acc + (long)d * 128;
  #pragma unroll
  for (int h = lane; h < 128; h += 32)
    atomicAdd(ar + h, w * xr[h]);
}

// ---------- bias + relu + bf16 ----------
__global__ void k_finalize(const float* __restrict__ acc, const float* __restrict__ bias,
                           bf16_t* __restrict__ hbf, int N) {
  long idx = (long)blockIdx.x * blockDim.x + threadIdx.x;
  if (idx >= (long)N * 128) return;
  float v = acc[idx] + bias[idx & 127];
  hbf[idx] = f2bf(v > 0.f ? v : 0.f);
}

// ---------- global mean pool ----------
__global__ void k_pool_init(float* __restrict__ pooled, float* __restrict__ cnt, int GH, int G) {
  int i = blockIdx.x * blockDim.x + threadIdx.x;
  if (i < GH) pooled[i] = 0.f;
  if (i < G)  cnt[i] = 0.f;
}

__global__ void k_pool(const float* __restrict__ x, const int* __restrict__ batch,
                       float* __restrict__ pooled, float* __restrict__ cnt, int N) {
  long idx = (long)blockIdx.x * blockDim.x + threadIdx.x;
  if (idx >= (long)N * 128) return;
  int i = (int)(idx >> 7), h = (int)(idx & 127);
  int g = batch[i];
  atomicAdd(pooled + (long)g * 128 + h, x[idx]);
  if (h == 0) atomicAdd(cnt + g, 1.f);
}

__global__ void k_pool_final(const float* __restrict__ pooled, const float* __restrict__ cnt,
                             bf16_t* __restrict__ pbf, int GH) {
  int i = blockIdx.x * blockDim.x + threadIdx.x;
  if (i >= GH) return;
  float c = cnt[i >> 7];
  pbf[i] = f2bf(pooled[i] / (c > 1.f ? c : 1.f));
}

// ---------- host ----------
extern "C" void kernel_launch(void* const* d_in, const int* in_sizes, int n_in,
                              void* d_out, int out_size, void* d_ws, size_t ws_size,
                              hipStream_t stream) {
  const int*   nodes   = (const int*)d_in[0];
  const int*   ei      = (const int*)d_in[1];
  const int*   batch   = (const int*)d_in[2];
  const float* emb     = (const float*)d_in[3];
  const float* gat_w   = (const float*)d_in[4];
  const float* att_src = (const float*)d_in[5];
  const float* att_dst = (const float*)d_in[6];
  const float* gat_b   = (const float*)d_in[7];
  const float* lin_w   = (const float*)d_in[8];
  const float* lin_b   = (const float*)d_in[9];
  const float* out_w   = (const float*)d_in[10];
  const float* out_b   = (const float*)d_in[11];
  float* out = (float*)d_out;

  const int N = in_sizes[0];
  const int E = in_sizes[1] / 2;
  const int H = 128;
  const int V = in_sizes[11];
  const int L = in_sizes[4] / (H * H);
  const int G = 64;
  const int EP = E + N;

  // workspace carve (256B aligned chunks)
  char* p = (char*)d_ws;
  auto carve = [&](size_t bytes) { void* r = (void*)p; p += (bytes + 255) & ~(size_t)255; return r; };
  float*  xw   = (float*)carve((size_t)N * H * 4);
  float*  acc  = (float*)carve((size_t)N * H * 4);   // reused as x_f32 after lin GEMM
  bf16_t* xbf  = (bf16_t*)carve((size_t)N * H * 2);
  bf16_t* hbf  = (bf16_t*)carve((size_t)N * H * 2);
  float*  es   = (float*)carve((size_t)N * 4);
  float*  edv  = (float*)carve((size_t)N * 4);
  float*  mx   = (float*)carve((size_t)N * 4);
  float*  dn   = (float*)carve((size_t)N * 4);
  bf16_t* wg   = (bf16_t*)carve((size_t)L * H * H * 2);  // transposed, per layer
  bf16_t* wl   = (bf16_t*)carve((size_t)L * H * H * 2);  // transposed, per layer
  bf16_t* wo   = (bf16_t*)carve((size_t)V * H * 2);      // out_w^T [V x 128]
  float*  pooled = (float*)carve((size_t)G * H * 4);
  float*  cnt    = (float*)carve((size_t)G * 4);
  bf16_t* pbf    = (bf16_t*)carve((size_t)G * H * 2);

  const int TB = 256;
  long nh = (long)N * H;

  // 1) weights -> transposed bf16 (one-time; L2-resident afterwards)
  for (int l = 0; l < L; l++) {
    dim3 gt(H / 32, H / 32);
    k_transpose_bf16<<<gt, TB, 0, stream>>>(gat_w + (size_t)l * H * H, wg + (size_t)l * H * H, H, H);
    k_transpose_bf16<<<gt, TB, 0, stream>>>(lin_w + (size_t)l * H * H, wl + (size_t)l * H * H, H, H);
  }
  {
    dim3 gt((V + 31) / 32, H / 32);
    k_transpose_bf16<<<gt, TB, 0, stream>>>(out_w, wo, H, V);
  }

  // 2) embedding gather -> bf16 activations
  k_embed<<<(int)((nh + TB - 1) / TB), TB, 0, stream>>>(nodes, emb, xbf, N);

  for (int l = 0; l < L; l++) {
    dim3 g1(N / 16, H / 128);
    // xw = x @ W   (bf16 WMMA, f32 accum)
    k_gemm_bf16<false, true, false><<<g1, TB, 0, stream>>>(
        xbf, wg + (size_t)l * H * H, nullptr, xw, nullptr, N, H);
    // per-node attention logits
    k_escore<<<(N + 7) / 8, TB, 0, stream>>>(xw, att_src + (size_t)l * H, att_dst + (size_t)l * H, es, edv, N);
    // segment softmax + scatter
    k_init_layer<<<(int)((nh + TB - 1) / TB), TB, 0, stream>>>(mx, dn, acc, N);
    k_edge_max<<<(EP + TB - 1) / TB, TB, 0, stream>>>(ei, E, EP, es, edv, mx);
    k_edge_sum<<<(EP + TB - 1) / TB, TB, 0, stream>>>(ei, E, EP, es, edv, mx, dn);
    k_edge_scatter<<<(EP + 7) / 8, TB, 0, stream>>>(ei, E, EP, es, edv, mx, dn, xw, acc);
    // h = relu(acc + gat_b) -> bf16
    k_finalize<<<(int)((nh + TB - 1) / TB), TB, 0, stream>>>(acc, gat_b + (size_t)l * H, hbf, N);
    // x = h @ lin_w + lin_b  (write f32 into acc, bf16 into xbf for next layer)
    k_gemm_bf16<true, true, true><<<g1, TB, 0, stream>>>(
        hbf, wl + (size_t)l * H * H, lin_b + (size_t)l * H, acc, xbf, N, H);
  }

  // 3) global mean pool
  k_pool_init<<<(G * H + G + TB - 1) / TB, TB, 0, stream>>>(pooled, cnt, G * H, G);
  k_pool<<<(int)((nh + TB - 1) / TB), TB, 0, stream>>>(acc, batch, pooled, cnt, N);
  k_pool_final<<<(G * H + TB - 1) / TB, TB, 0, stream>>>(pooled, cnt, pbf, G * H);

  // 4) logits = pooled @ out_w + out_b   [64 x 32000]
  dim3 g2(G / 16, V / 128);
  k_gemm_bf16<true, true, false><<<g2, TB, 0, stream>>>(pbf, wo, out_b, out, nullptr, G, V);
}
